// VectorQuantizer_11519102288293
// MI455X (gfx1250) — matrix operand order repaired and checked by reading, not compiled, not run on hardware
//
#include <hip/hip_runtime.h>
#include <stdint.h>

// CDNA5 wave32 WMMA types
typedef __attribute__((ext_vector_type(16))) __bf16 v16bf;
typedef __attribute__((ext_vector_type(8)))  float  v8f;
typedef __attribute__((ext_vector_type(8)))  int    v8i;

#define C_DIM        256   // channels (reduction dim)
#define ROWS_PER_WG  128   // tokens per workgroup (8 waves x 16 rows)
#define THREADS      256

// round-to-nearest-even f32 -> bf16 (inputs are normal floats)
__device__ __forceinline__ unsigned short f2bf(float f) {
  union { float f; unsigned int u; } cv; cv.f = f;
  unsigned int u = cv.u;
  u += 0x7FFFu + ((u >> 16) & 1u);
  return (unsigned short)(u >> 16);
}

// ---------------------------------------------------------------------------
// Prep: weight[K][C] f32 -> (a) bf16 pair-swizzled B layout in ws,
//                           (b) wnorm[k] = sum_c w^2 (f32)
// B pair layout (u16): idx = (((k>>4)*(C/2) + (c>>1))*16 + (k&15))*2 + (c&1)
// so each 16-code tile is a contiguous 8KB block whose u32 words are already
// in the ISA 32x16 bf16 B-matrix per-lane striping.
// ---------------------------------------------------------------------------
__global__ void vq_prep(const float* __restrict__ weight,
                        unsigned short* __restrict__ wb,
                        float* __restrict__ wnorm) {
  const int k = blockIdx.x;
  const int c = threadIdx.x;
  const float w = weight[(size_t)k * C_DIM + c];
  const int half_idx = ((((k >> 4) * (C_DIM / 2) + (c >> 1)) * 16) + (k & 15)) * 2 + (c & 1);
  wb[half_idx] = f2bf(w);

  __shared__ float red[THREADS];
  red[c] = w * w;
  __syncthreads();
  for (int s = THREADS / 2; s > 0; s >>= 1) {
    if (c < s) red[c] += red[c + s];
    __syncthreads();
  }
  if (c == 0) wnorm[k] = red[0];
}

// ---------------------------------------------------------------------------
// Main: per WG, 128 tokens vs all K codes.
// score_k = ||e_k||^2 - 2 * x.e_k   (||x||^2 is row-constant -> argmin-invariant)
// Double-buffered B tiles: fetch tile t+1 from L2 into registers while the
// 8-WMMA chain consumes tile t from LDS; one barrier per tile.
// ---------------------------------------------------------------------------
__global__ void __launch_bounds__(THREADS)
vq_main(const float* __restrict__ x,
        const float* __restrict__ weight,
        const unsigned int* __restrict__ wb,     // u32 pair words, tile-major
        const float* __restrict__ wnorm,
        float* __restrict__ out,
        int Ktiles) {
  __shared__ unsigned int ldsB[2 * 2048];        // 2 x 8KB double-buffered B tile
  __shared__ int sIdx[ROWS_PER_WG];

  const int tid  = threadIdx.x;
  const int wave = tid >> 5;
  const int lane = tid & 31;
  const int hi   = (lane >> 4) & 1;              // wave32 half
  const int l15  = lane & 15;

  const long nbase = (long)blockIdx.x * ROWS_PER_WG;   // first token of WG
  const long b   = nbase >> 14;                  // 16384 tokens per batch elem
  const long thw = nbase & 16383;

  // x element (token n, chan c) lives at ((b*C + c)*16384 + n%16384):
  // consecutive tokens are contiguous for fixed c -> coalesced A loads.
  const float* xr = x + (b * C_DIM) * 16384 + thw + wave * 16 + l15;

  // --- Load A fragments once (ISA 16-bit A 16x32 layout), reuse for all tiles
  v16bf afrag[8];
  #pragma unroll
  for (int ks = 0; ks < 8; ++ks) {
    v8i wds;
    #pragma unroll
    for (int v = 0; v < 8; ++v) {
      const int c = 32 * ks + 2 * ((v & 3) + ((v >> 2) << 3) + (hi << 2));
      const unsigned int h0 = f2bf(xr[(long)c * 16384]);
      const unsigned int h1 = f2bf(xr[(long)(c + 1) * 16384]);
      wds[v] = (int)(h0 | (h1 << 16));
    }
    afrag[ks] = __builtin_bit_cast(v16bf, wds);
  }

  float best[8];
  int   bidx[8];
  #pragma unroll
  for (int j = 0; j < 8; ++j) { best[j] = 3.4e38f; bidx[j] = 0; }

  // Preload tile 0 into registers
  unsigned int rbuf[8];
  #pragma unroll
  for (int j = 0; j < 8; ++j) rbuf[j] = wb[tid + j * 256];

  for (int tile = 0; tile < Ktiles; ++tile) {
    unsigned int* dst = ldsB + (tile & 1) * 2048;
    #pragma unroll
    for (int j = 0; j < 8; ++j) dst[tid + j * 256] = rbuf[j];
    __syncthreads();                             // tile ready; prior buf reads done

    // Fetch next tile from L2 into registers (overlaps with WMMA chain below)
    if (tile + 1 < Ktiles) {
      const unsigned int* src = wb + (size_t)(tile + 1) * 2048;
      #pragma unroll
      for (int j = 0; j < 8; ++j) rbuf[j] = src[tid + j * 256];
    }
    if (tile + 2 < Ktiles)                       // gfx1250 global_prefetch_b8: warm L2
      __builtin_prefetch(wb + (size_t)(tile + 2) * 2048 + tid * 8, 0, 0);

    v8f acc = {};
    #pragma unroll
    for (int ks = 0; ks < 8; ++ks) {
      v8i bw;
      #pragma unroll
      for (int v = 0; v < 8; ++v)
        bw[v] = (int)dst[(ks * 16 + (v & 3) + ((v >> 2) << 3) + (hi << 2)) * 16 + l15];
      const v16bf bfrag = __builtin_bit_cast(v16bf, bw);
      acc = __builtin_amdgcn_wmma_f32_16x16x32_bf16(
          false, afrag[ks], false, bfrag, (short)0, acc, false, false);
    }

    // acc VGPR j -> (row = wave*16 + j + hi*8, code n = tile*16 + l15)
    const int n = tile * 16 + l15;
    const float wn = wnorm[n];
    #pragma unroll
    for (int j = 0; j < 8; ++j) {
      const float s = wn - 2.0f * acc[j];
      if (s < best[j]) { best[j] = s; bidx[j] = n; }   // strict < keeps lowest n on ties
    }
  }

  // argmin across the 16 lanes holding a row's columns (xor<16 stays in-half)
  #pragma unroll
  for (int j = 0; j < 8; ++j) {
    #pragma unroll
    for (int m = 1; m < 16; m <<= 1) {
      const float ov = __shfl_xor(best[j], m, 32);
      const int   oi = __shfl_xor(bidx[j], m, 32);
      if (ov < best[j] || (ov == best[j] && oi < bidx[j])) { best[j] = ov; bidx[j] = oi; }
    }
    if (l15 == 0) sIdx[wave * 16 + j + hi * 8] = bidx[j];
  }
  __syncthreads();

  // gather f32 codebook rows -> out (coalesced over c)
  for (int i = tid; i < ROWS_PER_WG * C_DIM; i += THREADS) {
    const int r = i >> 8;       // / C_DIM
    const int c = i & (C_DIM - 1);
    out[(nbase + r) * C_DIM + c] = weight[(size_t)sIdx[r] * C_DIM + c];
  }
}

// ---------------------------------------------------------------------------
extern "C" void kernel_launch(void* const* d_in, const int* in_sizes, int n_in,
                              void* d_out, int out_size, void* d_ws, size_t ws_size,
                              hipStream_t stream) {
  const float* x      = (const float*)d_in[0];
  const float* weight = (const float*)d_in[1];
  const int K    = in_sizes[1] / C_DIM;   // 1024 codes
  const int Ntok = in_sizes[0] / C_DIM;   // 65536 tokens

  unsigned short* wb = (unsigned short*)d_ws;                       // K*C bf16 = 512KB
  float* wnorm = (float*)((char*)d_ws + (size_t)K * C_DIM * 2);     // K f32

  vq_prep<<<K, THREADS, 0, stream>>>(weight, wb, wnorm);
  vq_main<<<Ntok / ROWS_PER_WG, THREADS, 0, stream>>>(
      x, weight, (const unsigned int*)wb, wnorm, (float*)d_out, K / 16);
}
